// BiDirectionalAddBlock_73581379715114
// MI455X (gfx1250) — compile-verified, matrix operand order checked by reading, not run on hardware
//
#include <hip/hip_runtime.h>

// ---------------------------------------------------------------------------
// Bidirectional Mamba block for MI455X (gfx1250, wave32, WMMA).
// GEMMs run on v_wmma_f32_16x16x32_bf16 (fp32 accum); scan/conv/softplus fp32.
// GEMM: 128x64 block tile, 8 waves, each wave = 16 rows x 64 cols
// (4 accumulators -> 4 WMMAs per K-step per wave), global_prefetch of the
// next K slab overlapped with compute.
// ---------------------------------------------------------------------------

#define D_MODEL 768
#define D_INNER 1536
#define DT_RANK 48
#define D_STATE 16
#define D_CONV  4
#define BATCH   2
#define SEQ     1024
#define ROWS    (BATCH * SEQ)           // 2048
#define N_XZ    (2 * D_INNER)           // 3072
#define N_XP    (DT_RANK + 2 * D_STATE) // 80

typedef __attribute__((ext_vector_type(16))) __bf16 v16bf;
typedef __attribute__((ext_vector_type(8)))  __bf16 v8bf;
typedef __attribute__((ext_vector_type(8)))  float  v8f;

union FragBF {
    v16bf v;
    v8bf  h[2];
};

__device__ __forceinline__ unsigned short f2bf(float f) {
    unsigned int u = __float_as_uint(f);
    u += 0x7FFFu + ((u >> 16) & 1u);   // round-to-nearest-even
    return (unsigned short)(u >> 16);
}

__device__ __forceinline__ float silu_f(float v) {
    return v * (1.0f / (1.0f + __expf(-v)));
}

// ---------------------------------------------------------------------------
// LayerNorm -> bf16 (forward + row-reversed copies) + residual passthrough.
// One block per row (768 elements, 256 threads).
// ---------------------------------------------------------------------------
__global__ __launch_bounds__(256) void ln_kernel(
    const float* __restrict__ x, const float* __restrict__ gamma,
    const float* __restrict__ beta,
    unsigned short* __restrict__ xn_fwd, unsigned short* __restrict__ xn_rev,
    float* __restrict__ resid)
{
    const int r   = blockIdx.x;           // 0..2047
    const int b   = r >> 10;
    const int l   = r & 1023;
    const int tid = threadIdx.x;
    __shared__ float red[256];

    const float* xr = x + (size_t)r * D_MODEL;

    float s = 0.f;
    for (int i = tid; i < D_MODEL; i += 256) s += xr[i];
    red[tid] = s;
    __syncthreads();
    for (int st = 128; st > 0; st >>= 1) {
        if (tid < st) red[tid] += red[tid + st];
        __syncthreads();
    }
    const float mu = red[0] * (1.0f / D_MODEL);
    __syncthreads();

    float s2 = 0.f;
    for (int i = tid; i < D_MODEL; i += 256) {
        float d = xr[i] - mu;
        s2 += d * d;
    }
    red[tid] = s2;
    __syncthreads();
    for (int st = 128; st > 0; st >>= 1) {
        if (tid < st) red[tid] += red[tid + st];
        __syncthreads();
    }
    const float rstd = rsqrtf(red[0] * (1.0f / D_MODEL) + 1e-5f);

    const int rr = b * SEQ + (SEQ - 1 - l);    // reversed row index
    for (int i = tid; i < D_MODEL; i += 256) {
        float v  = xr[i];
        float yn = (v - mu) * rstd * gamma[i] + beta[i];
        unsigned short bf = f2bf(yn);
        xn_fwd[(size_t)r  * D_MODEL + i] = bf;
        xn_rev[(size_t)rr * D_MODEL + i] = bf;
        resid[(size_t)r * D_MODEL + i]   = v;
    }
}

// ---------------------------------------------------------------------------
// fp32 -> bf16 conversion (weights)
// ---------------------------------------------------------------------------
__global__ __launch_bounds__(256) void cvt_bf16_kernel(
    const float* __restrict__ src, unsigned short* __restrict__ dst, int n)
{
    int i = blockIdx.x * 256 + threadIdx.x;
    if (i < n) dst[i] = f2bf(src[i]);
}

// ---------------------------------------------------------------------------
// bf16 WMMA GEMM: C[M,N] (+)= A[M,K] * W[N,K]^T   (A,W bf16 row-major)
// Block = 256 threads (8 waves). Block tile 128x64, BK=32.
// Wave w owns rows [w*16, w*16+16) x all 64 cols -> 4 accumulators,
// 4 v_wmma per K-step with a single A fragment (max A reuse).
// ---------------------------------------------------------------------------
__global__ __launch_bounds__(256) void gemm_bf16_kernel(
    const unsigned short* __restrict__ A, const unsigned short* __restrict__ Wt,
    float* __restrict__ C, int M, int N, int K, int accumulate)
{
    __shared__ __align__(16) unsigned short As[128 * 32];  // 8 KB
    __shared__ __align__(16) unsigned short Bs[64 * 32];   // 4 KB

    const int tid  = threadIdx.x;
    const int bm   = blockIdx.y * 128;
    const int bn   = blockIdx.x * 64;
    const int wave = tid >> 5;        // 0..7 -> 16-row strip
    const int lane = tid & 31;
    const int g    = lane >> 4;       // half-wave group
    const int ml   = lane & 15;

    // staging coords: 16B chunks of a 32-wide bf16 k-slab
    const int lrow = tid >> 2;        // 0..63
    const int lcol = (tid & 3) * 8;   // 0,8,16,24

    v8f acc[4] = {v8f{}, v8f{}, v8f{}, v8f{}};

    for (int k0 = 0; k0 < K; k0 += 32) {
        // ---- stage A slab: 128 rows x 32 k (two 16B chunks per thread)
        *(uint4*)(As + lrow * 32 + lcol) =
            *(const uint4*)(A + (size_t)(bm + lrow) * K + k0 + lcol);
        *(uint4*)(As + (lrow + 64) * 32 + lcol) =
            *(const uint4*)(A + (size_t)(bm + lrow + 64) * K + k0 + lcol);
        // ---- stage W slab: 64 n-rows x 32 k, zero-pad past N (N=80 case)
        {
            int n = bn + lrow;
            uint4 val;
            if (n < N) {
                val = *(const uint4*)(Wt + (size_t)n * K + k0 + lcol);
            } else {
                val.x = 0u; val.y = 0u; val.z = 0u; val.w = 0u;
            }
            *(uint4*)(Bs + lrow * 32 + lcol) = val;
        }
        // ---- prefetch next K slab while this one computes
        if (k0 + 32 < K) {
            __builtin_prefetch(A + (size_t)(bm + lrow) * K + k0 + 32 + lcol, 0, 1);
            __builtin_prefetch(A + (size_t)(bm + lrow + 64) * K + k0 + 32 + lcol, 0, 1);
            if (bn + lrow < N)
                __builtin_prefetch(Wt + (size_t)(bn + lrow) * K + k0 + 32 + lcol, 0, 1);
        }
        __syncthreads();

        // A fragment: lane (g,ml) holds A[m][g*8..+7] and A[m][16+g*8..+7]
        FragBF af;
        const __bf16* ap = (const __bf16*)(As + (wave * 16 + ml) * 32);
        af.h[0] = *(const v8bf*)(ap + g * 8);
        af.h[1] = *(const v8bf*)(ap + 16 + g * 8);

        #pragma unroll
        for (int tn = 0; tn < 4; ++tn) {
            FragBF bf;
            const __bf16* bp = (const __bf16*)(Bs + (tn * 16 + ml) * 32);
            bf.h[0] = *(const v8bf*)(bp + g * 8);
            bf.h[1] = *(const v8bf*)(bp + 16 + g * 8);
            acc[tn] = __builtin_amdgcn_wmma_f32_16x16x32_bf16(
                false, af.v, false, bf.v, (short)0, acc[tn], false, false);
        }
        __syncthreads();
    }

    // C/D layout: lane (g,ml): element v -> row (v + 8*g), col ml
    #pragma unroll
    for (int tn = 0; tn < 4; ++tn) {
        int n = bn + tn * 16 + ml;
        if (n < N) {
            #pragma unroll
            for (int v = 0; v < 8; ++v) {
                int m = bm + wave * 16 + v + 8 * g;
                float o = acc[tn][v];
                if (accumulate) o += C[(size_t)m * N + n];
                C[(size_t)m * N + n] = o;
            }
        }
    }
}

// ---------------------------------------------------------------------------
// Causal depthwise conv (k=4) + SiLU on xi = xz[:, :D_INNER]
// ---------------------------------------------------------------------------
__global__ __launch_bounds__(256) void conv_silu_kernel(
    const float* __restrict__ xz, const float* __restrict__ convw,
    const float* __restrict__ convb,
    float* __restrict__ xc, unsigned short* __restrict__ xcbf)
{
    int idx = blockIdx.x * 256 + threadIdx.x;
    if (idx >= ROWS * D_INNER) return;
    int r = idx / D_INNER;
    int d = idx - r * D_INNER;
    int b = r >> 10;
    int l = r & 1023;

    float acc = convb[d];
    #pragma unroll
    for (int k = 0; k < D_CONV; ++k) {
        int ll = l - (D_CONV - 1) + k;
        if (ll >= 0)
            acc += xz[(size_t)(b * SEQ + ll) * N_XZ + d] * convw[d * D_CONV + k];
    }
    float s = silu_f(acc);
    xc[idx]   = s;
    xcbf[idx] = f2bf(s);
}

// ---------------------------------------------------------------------------
// delta = softplus(dtp @ dtW^T + dtb), dtp = dbc[:, :48]
// ---------------------------------------------------------------------------
__global__ __launch_bounds__(256) void delta_kernel(
    const float* __restrict__ dbc, const float* __restrict__ dtW,
    const float* __restrict__ dtb, float* __restrict__ delta)
{
    int idx = blockIdx.x * 256 + threadIdx.x;
    if (idx >= ROWS * D_INNER) return;
    int r = idx / D_INNER;
    int d = idx - r * D_INNER;

    float s = dtb[d];
    const float* row = dbc + (size_t)r * N_XP;
    const float* w   = dtW + (size_t)d * DT_RANK;
    #pragma unroll 8
    for (int j = 0; j < DT_RANK; ++j) s += row[j] * w[j];
    float sp = (s > 20.f) ? s : log1pf(__expf(s));
    delta[idx] = sp;
}

// ---------------------------------------------------------------------------
// Selective scan: channel-parallel (one thread per (b,d)), serial over t.
// B/C (32 floats per (b,t)) broadcast via LDS to 256 channels per block.
// ---------------------------------------------------------------------------
__global__ __launch_bounds__(256) void scan_kernel(
    const float* __restrict__ dbc, const float* __restrict__ delta,
    const float* __restrict__ xc,  const float* __restrict__ Alog,
    float* __restrict__ ys)
{
    const int blocks_per_b = D_INNER / 256;   // 6
    const int b = blockIdx.x / blocks_per_b;
    const int d = (blockIdx.x % blocks_per_b) * 256 + threadIdx.x;

    __shared__ float BC[2 * D_STATE];

    float Arow[D_STATE];
    #pragma unroll
    for (int n = 0; n < D_STATE; ++n)
        Arow[n] = -__expf(Alog[(size_t)d * D_STATE + n]);

    float h[D_STATE];
    #pragma unroll
    for (int n = 0; n < D_STATE; ++n) h[n] = 0.f;

    for (int t = 0; t < SEQ; ++t) {
        const int r = b * SEQ + t;
        if (threadIdx.x < 2 * D_STATE)
            BC[threadIdx.x] = dbc[(size_t)r * N_XP + DT_RANK + threadIdx.x];
        __syncthreads();

        const float dlt = delta[(size_t)r * D_INNER + d];
        const float dx  = dlt * xc[(size_t)r * D_INNER + d];
        float y = 0.f;
        #pragma unroll
        for (int n = 0; n < D_STATE; ++n) {
            float dA = __expf(dlt * Arow[n]);
            h[n] = dA * h[n] + BC[n] * dx;
            y   += h[n] * BC[D_STATE + n];
        }
        ys[(size_t)r * D_INNER + d] = y;
        __syncthreads();
    }
}

// ---------------------------------------------------------------------------
// y = (ys + xc*Dsk) * silu(z)  -> bf16 for the output GEMM
// ---------------------------------------------------------------------------
__global__ __launch_bounds__(256) void gate_kernel(
    const float* __restrict__ ys, const float* __restrict__ xc,
    const float* __restrict__ xz, const float* __restrict__ Dsk,
    unsigned short* __restrict__ ybf)
{
    int idx = blockIdx.x * 256 + threadIdx.x;
    if (idx >= ROWS * D_INNER) return;
    int r = idx / D_INNER;
    int d = idx - r * D_INNER;

    float z = xz[(size_t)r * N_XZ + D_INNER + d];
    float y = (ys[idx] + xc[idx] * Dsk[d]) * silu_f(z);
    ybf[idx] = f2bf(y);
}

// ---------------------------------------------------------------------------
extern "C" void kernel_launch(void* const* d_in, const int* in_sizes, int n_in,
                              void* d_out, int out_size, void* d_ws, size_t ws_size,
                              hipStream_t stream)
{
    (void)in_sizes; (void)n_in; (void)out_size; (void)ws_size;

    const float* x    = (const float*)d_in[0];
    const float* ln_g = (const float*)d_in[1];
    const float* ln_b = (const float*)d_in[2];

    float* hidden = (float*)d_out;                              // (B,L,768)
    float* resid  = (float*)d_out + (size_t)ROWS * D_MODEL;     // (B,L,768)

    // ---- workspace carve-out --------------------------------------------
    char*  ws  = (char*)d_ws;
    size_t off = 0;
    auto carve = [&](size_t bytes) -> void* {
        void* p = ws + off;
        off = (off + bytes + 255) & ~(size_t)255;
        return p;
    };
    unsigned short* xnbf0   = (unsigned short*)carve((size_t)ROWS * D_MODEL * 2);
    unsigned short* xnbf1   = (unsigned short*)carve((size_t)ROWS * D_MODEL * 2);
    unsigned short* w_in_bf = (unsigned short*)carve((size_t)N_XZ * D_MODEL * 2);
    unsigned short* w_xp_bf = (unsigned short*)carve((size_t)N_XP * D_INNER * 2);
    unsigned short* w_out_bf= (unsigned short*)carve((size_t)D_MODEL * D_INNER * 2);
    float*          xz      = (float*)carve((size_t)ROWS * N_XZ * 4);
    float*          xc      = (float*)carve((size_t)ROWS * D_INNER * 4);
    unsigned short* xcbf    = (unsigned short*)carve((size_t)ROWS * D_INNER * 2);
    float*          dbc     = (float*)carve((size_t)ROWS * N_XP * 4);
    float*          delta   = (float*)carve((size_t)ROWS * D_INNER * 4);
    float*          ys      = (float*)carve((size_t)ROWS * D_INNER * 4);
    unsigned short* ybf     = (unsigned short*)carve((size_t)ROWS * D_INNER * 2);

    // ---- layernorm + residual + bf16 fwd/rev copies ---------------------
    ln_kernel<<<ROWS, 256, 0, stream>>>(x, ln_g, ln_b, xnbf0, xnbf1, resid);

    const int ew_blocks = (ROWS * D_INNER + 255) / 256;

    for (int m = 0; m < 2; ++m) {
        const int base = 3 + m * 9;
        const float* inW   = (const float*)d_in[base + 0];
        const float* convw = (const float*)d_in[base + 1];
        const float* convb = (const float*)d_in[base + 2];
        const float* xpW   = (const float*)d_in[base + 3];
        const float* dtW   = (const float*)d_in[base + 4];
        const float* dtb   = (const float*)d_in[base + 5];
        const float* Alog  = (const float*)d_in[base + 6];
        const float* Dsk   = (const float*)d_in[base + 7];
        const float* outW  = (const float*)d_in[base + 8];

        // Convert this direction's GEMM weights to bf16
        cvt_bf16_kernel<<<(N_XZ * D_MODEL + 255) / 256, 256, 0, stream>>>(
            inW, w_in_bf, N_XZ * D_MODEL);
        cvt_bf16_kernel<<<(N_XP * D_INNER + 255) / 256, 256, 0, stream>>>(
            xpW, w_xp_bf, N_XP * D_INNER);
        cvt_bf16_kernel<<<(D_MODEL * D_INNER + 255) / 256, 256, 0, stream>>>(
            outW, w_out_bf, D_MODEL * D_INNER);

        const unsigned short* xn = (m == 0) ? xnbf0 : xnbf1;

        // xz = xn @ inW^T     (2048 x 3072, K=768)
        gemm_bf16_kernel<<<dim3(N_XZ / 64, ROWS / 128), 256, 0, stream>>>(
            xn, w_in_bf, xz, ROWS, N_XZ, D_MODEL, 0);

        // causal conv + SiLU
        conv_silu_kernel<<<ew_blocks, 256, 0, stream>>>(xz, convw, convb, xc, xcbf);

        // dbc = xc @ xpW^T    (2048 x 80, K=1536)
        gemm_bf16_kernel<<<dim3((N_XP + 63) / 64, ROWS / 128), 256, 0, stream>>>(
            xcbf, w_xp_bf, dbc, ROWS, N_XP, D_INNER, 0);

        // delta = softplus(dtp @ dtW^T + dtb)
        delta_kernel<<<ew_blocks, 256, 0, stream>>>(dbc, dtW, dtb, delta);

        // selective scan
        scan_kernel<<<BATCH * (D_INNER / 256), 256, 0, stream>>>(
            dbc, delta, xc, Alog, ys);

        // gate: y = (ys + xc*Dsk) * silu(z)
        gate_kernel<<<ew_blocks, 256, 0, stream>>>(ys, xc, xz, Dsk, ybf);

        // hidden (+)= y @ outW^T   (2048 x 768, K=1536); m==1 accumulates
        gemm_bf16_kernel<<<dim3(D_MODEL / 64, ROWS / 128), 256, 0, stream>>>(
            ybf, w_out_bf, hidden, ROWS, D_MODEL, D_INNER, m);
    }
}
